// BaseConvBlock_19224273617327
// MI455X (gfx1250) — matrix operand order repaired and checked by reading, not compiled, not run on hardware
//
#include <hip/hip_runtime.h>
#include <stdint.h>

// ---------------- types for WMMA ----------------
typedef __attribute__((ext_vector_type(16))) __bf16   v16bf;
typedef __attribute__((ext_vector_type(8)))  float    v8f;
typedef __attribute__((ext_vector_type(8)))  unsigned int v8u;

#define KVOL 27
#define CIN  32
#define COUT 32
#define WPACK_DWORDS (KVOL * 2 * 32 * 8)   // [k][half][lane][8 dwords] = 13824

// round-to-nearest-even fp32 -> bf16, packed pair
__device__ __forceinline__ uint32_t pack_bf16x2(float a, float b) {
    uint32_t ua = __float_as_uint(a), ub = __float_as_uint(b);
    ua = (ua + 0x7FFFu + ((ua >> 16) & 1u)) >> 16;
    ub = (ub + 0x7FFFu + ((ub >> 16) & 1u)) >> 16;
    return (ua & 0xFFFFu) | (ub << 16);
}

// ---------------- kernel 1: pack W into bf16 B-fragment layout ----------------
// wpack[i], i = (k*2 + h)*256 + lane*8 + d
// dword d of lane holds K rows (2d, 2d+1) + (lane>=16 ? 16 : 0), column (lane&15) + 16h
__global__ __launch_bounds__(256) void pack_w_kernel(const float* __restrict__ W,
                                                     uint32_t* __restrict__ wpack) {
    for (int i = threadIdx.x; i < WPACK_DWORDS; i += 256) {
        const int d    = i & 7;
        const int lane = (i >> 3) & 31;
        const int h    = (i >> 8) & 1;
        const int k    = i >> 9;
        const int col  = (lane & 15) + 16 * h;
        const int kr   = 2 * d + ((lane >> 4) << 4);
        const float f0 = W[k * (CIN * COUT) + kr * COUT + col];
        const float f1 = W[k * (CIN * COUT) + (kr + 1) * COUT + col];
        wpack[i] = pack_bf16x2(f0, f1);
    }
}

// ---------------- kernel 2: feats fp32 -> bf16 ----------------
__global__ __launch_bounds__(256) void feats_to_bf16_kernel(const float2* __restrict__ x,
                                                            uint32_t* __restrict__ out,
                                                            size_t n2) {
    size_t i = (size_t)blockIdx.x * blockDim.x + threadIdx.x;
    const size_t stride = (size_t)gridDim.x * blockDim.x;
    for (; i < n2; i += stride) {
        float2 v = x[i];
        out[i] = pack_bf16x2(v.x, v.y);
    }
}

// ---------------- kernel 3: gather + WMMA conv + partial BN stats ----------------
__global__ __launch_bounds__(256) void spconv_wmma_kernel(
    const uint16_t* __restrict__ feats16, const uint32_t* __restrict__ wpack,
    const int* __restrict__ nbr_idx, const unsigned char* __restrict__ nbr_mask,
    float* __restrict__ y, float* __restrict__ partials, int Nv)
{
    __shared__ uint32_t sW[WPACK_DWORDS];
    __shared__ float sPart[8][64];

    const int tid = threadIdx.x;
    for (int i = tid; i < WPACK_DWORDS; i += 256) sW[i] = wpack[i];
    __syncthreads();

    const int wave = tid >> 5;
    const int lane = tid & 31;
    const int tile = blockIdx.x * 8 + wave;
    const int tileBase = tile * 16;

    v8f c0 = {};
    v8f c1 = {};

    if (tileBase + 16 <= Nv) {                 // wave-uniform: EXEC all-1s inside
        const int  row  = tileBase + (lane & 15);
        const int  sel  = lane >> 4;           // 0: K0-15, 1: K16-31
        const long ibas = (long)row * KVOL;

        for (int k = 0; k < KVOL; ++k) {
            const int  idx = nbr_idx[ibas + k];
            const bool m   = nbr_mask[ibas + k] != 0;

            // gather one bf16 row (64B) split by lane half h into K 16h..16h+15
            const uint4* p = (const uint4*)(feats16 + (size_t)idx * CIN);
            uint4 alo = make_uint4(0u, 0u, 0u, 0u);
            uint4 ahi = make_uint4(0u, 0u, 0u, 0u);
            if (m) { alo = p[2 * sel]; ahi = p[2 * sel + 1]; }   // predicated loads only

            v8u au = { alo.x, alo.y, alo.z, alo.w, ahi.x, ahi.y, ahi.z, ahi.w };
            v16bf A = __builtin_bit_cast(v16bf, au);

            // B fragments for the two COUT halves, from LDS
            const uint32_t* wb = &sW[k * 512 + lane * 8];
            uint4 b0lo = *(const uint4*)(wb);
            uint4 b0hi = *(const uint4*)(wb + 4);
            uint4 b1lo = *(const uint4*)(wb + 256);
            uint4 b1hi = *(const uint4*)(wb + 260);
            v8u b0u = { b0lo.x, b0lo.y, b0lo.z, b0lo.w, b0hi.x, b0hi.y, b0hi.z, b0hi.w };
            v8u b1u = { b1lo.x, b1lo.y, b1lo.z, b1lo.w, b1hi.x, b1hi.y, b1hi.z, b1hi.w };
            v16bf B0 = __builtin_bit_cast(v16bf, b0u);
            v16bf B1 = __builtin_bit_cast(v16bf, b1u);

            c0 = __builtin_amdgcn_wmma_f32_16x16x32_bf16(false, A, false, B0,
                                                         (short)0, c0, false, false);
            c1 = __builtin_amdgcn_wmma_f32_16x16x32_bf16(false, A, false, B1,
                                                         (short)0, c1, false, false);
        }

        // store conv output (C layout: lanes 0-15 -> M=r, lanes 16-31 -> M=8+r; N = lane&15)
        const int n     = lane & 15;
        const int mbase = sel * 8;
        for (int r = 0; r < 8; ++r) {
            const size_t rowg = (size_t)(tileBase + mbase + r) * COUT;
            y[rowg + n]      = c0[r];
            y[rowg + 16 + n] = c1[r];
        }
    }

    // deterministic per-block column sum / sumsq (zeros for inactive waves)
    float s0 = 0.f, q0 = 0.f, s1 = 0.f, q1 = 0.f;
    for (int r = 0; r < 8; ++r) {
        s0 += c0[r]; q0 += c0[r] * c0[r];
        s1 += c1[r]; q1 += c1[r] * c1[r];
    }
    s0 += __shfl_xor(s0, 16, 32);
    q0 += __shfl_xor(q0, 16, 32);
    s1 += __shfl_xor(s1, 16, 32);
    q1 += __shfl_xor(q1, 16, 32);
    if (lane < 16) {
        sPart[wave][lane]      = s0;
        sPart[wave][16 + lane] = s1;
        sPart[wave][32 + lane] = q0;
        sPart[wave][48 + lane] = q1;
    }
    __syncthreads();
    if (tid < 64) {
        float t = 0.f;
        for (int w = 0; w < 8; ++w) t += sPart[w][tid];
        partials[(size_t)blockIdx.x * 64 + tid] = t;
    }
}

// ---------------- kernel 4: reduce partials -> per-channel scale/shift ----------------
__global__ __launch_bounds__(64) void bn_stats_kernel(const float* __restrict__ partials,
                                                      int nblocks,
                                                      const float* __restrict__ gamma,
                                                      const float* __restrict__ beta,
                                                      float* __restrict__ ss, float invN) {
    __shared__ float acc[64];
    const int t = threadIdx.x;
    float s = 0.f;
#pragma unroll 8
    for (int b = 0; b < nblocks; ++b) s += partials[(size_t)b * 64 + t];
    acc[t] = s;
    __syncthreads();
    if (t < 32) {
        const float mean = acc[t] * invN;
        const float var  = acc[32 + t] * invN - mean * mean;
        const float sc   = gamma[t] * rsqrtf(var + 1e-5f);
        ss[t]      = sc;
        ss[32 + t] = beta[t] - mean * sc;
    }
}

// ---------------- kernel 5: normalize + ReLU in place ----------------
__global__ __launch_bounds__(256) void bn_relu_kernel(float* __restrict__ y,
                                                      const float* __restrict__ ss,
                                                      size_t total) {
    __shared__ float s_ss[64];
    if (threadIdx.x < 64) s_ss[threadIdx.x] = ss[threadIdx.x];
    __syncthreads();
    size_t i = (size_t)blockIdx.x * blockDim.x + threadIdx.x;
    const size_t stride = (size_t)gridDim.x * blockDim.x;
    for (; i < total; i += stride) {
        const int c = (int)(i & 31);
        float v = y[i] * s_ss[c] + s_ss[32 + c];
        y[i] = v > 0.f ? v : 0.f;
    }
}

// ---------------- host launcher ----------------
extern "C" void kernel_launch(void* const* d_in, const int* in_sizes, int n_in,
                              void* d_out, int out_size, void* d_ws, size_t ws_size,
                              hipStream_t stream) {
    const float*         feats    = (const float*)d_in[0];
    const float*         W        = (const float*)d_in[1];
    const float*         gamma    = (const float*)d_in[2];
    const float*         beta     = (const float*)d_in[3];
    const int*           nbr_idx  = (const int*)d_in[4];
    const unsigned char* nbr_mask = (const unsigned char*)d_in[5];
    float*               y        = (float*)d_out;

    const int N = in_sizes[0] / CIN;

    // workspace carve-up
    char*  ws  = (char*)d_ws;
    size_t off = 0;
    uint16_t* feats16 = (uint16_t*)(ws + off);
    off += (size_t)N * CIN * sizeof(uint16_t);
    off = (off + 255) & ~(size_t)255;
    uint32_t* wpack = (uint32_t*)(ws + off);
    off += (size_t)WPACK_DWORDS * 4;
    off = (off + 255) & ~(size_t)255;

    const int ntiles  = N / 16;
    const int nblocks = (ntiles + 7) / 8;
    float* partials = (float*)(ws + off);
    off += (size_t)nblocks * 64 * sizeof(float);
    off = (off + 255) & ~(size_t)255;
    float* ss = (float*)(ws + off);

    // 1) pack weights to bf16 fragment layout
    pack_w_kernel<<<1, 256, 0, stream>>>(W, wpack);

    // 2) feats fp32 -> bf16
    {
        const size_t n2 = (size_t)N * CIN / 2;
        int grid = (int)((n2 + 255) / 256);
        if (grid > 65520) grid = 65520;
        feats_to_bf16_kernel<<<grid, 256, 0, stream>>>((const float2*)feats,
                                                       (uint32_t*)feats16, n2);
    }

    // 3) gather + WMMA conv, write y + per-block stats partials
    spconv_wmma_kernel<<<nblocks, 256, 0, stream>>>(feats16, wpack, nbr_idx, nbr_mask,
                                                    y, partials, N);

    // 4) reduce stats -> scale/shift
    bn_stats_kernel<<<1, 64, 0, stream>>>(partials, nblocks, gamma, beta, ss,
                                          1.0f / (float)N);

    // 5) normalize + ReLU in place
    {
        const size_t total = (size_t)N * COUT;
        bn_relu_kernel<<<8192, 256, 0, stream>>>(y, ss, total);
    }
}
